// VesselContinuityModule_85856396247191
// MI455X (gfx1250) — matrix operand order repaired and checked by reading, not compiled, not run on hardware
//
#include <hip/hip_runtime.h>
#include <math.h>

typedef __attribute__((ext_vector_type(16))) _Float16 v16h;
typedef __attribute__((ext_vector_type(8)))  _Float16 v8h;
typedef __attribute__((ext_vector_type(8)))  float    v8f;
typedef __attribute__((ext_vector_type(4)))  float    v4f;

#define N_NODES   50000
#define FDIM      64
#define N_EDGES   800000
#define NTILES    (N_EDGES / 16)           // 50000, exact
#define WPB       8                        // waves per block (256 threads, wave32)
#define EBLOCKS   625
#define TOT_WAVES (EBLOCKS * WPB)          // 5000 -> 10 tiles per wave exactly
#define UPD_ELEMS (N_NODES * FDIM)         // 3,200,000
#define SCORE_OFF UPD_ELEMS
#define AVG_OFF   (UPD_ELEMS + N_EDGES)    // 4,000,000

__device__ __forceinline__ v8h load8_cvt(const float* p) {
    const v4f* q = (const v4f*)p;
    v4f f0 = q[0], f1 = q[1];
    v8f f = __builtin_shufflevector(f0, f1, 0,1,2,3,4,5,6,7);
    return __builtin_convertvector(f, v8h);
}

__global__ void __launch_bounds__(256) vessel_init_kernel(const float* __restrict__ feats,
                                                          float* __restrict__ out,
                                                          float* __restrict__ ws) {
    long long i = (long long)blockIdx.x * blockDim.x + threadIdx.x;
    const long long n4 = UPD_ELEMS / 4;
    const v4f* s = (const v4f*)feats;
    v4f* d = (v4f*)out;
    for (long long k = i; k < n4; k += (long long)gridDim.x * blockDim.x) d[k] = s[k];
    if (i == 0) { ws[0] = 0.0f; ws[1] = 0.0f; }
}

__global__ void __launch_bounds__(256) vessel_edge_kernel(
    const float* __restrict__ feats, const long long* __restrict__ ei,
    const float* __restrict__ W1, const float* __restrict__ b1,
    const float* __restrict__ W2, const float* __restrict__ b2,
    const float* __restrict__ W3, const float* __restrict__ b3,
    float* __restrict__ out, float* __restrict__ ws)
{
    // LDS: weights pre-permuted into WMMA B-fragment lane order.
    __shared__ __align__(32) _Float16 w1B[4 * 4 * 32 * 16];   // [ntile][kchunk][lane][16]  16 KB
    __shared__ __align__(32) _Float16 w2B[2 * 2 * 32 * 16];   //                            4 KB
    __shared__ float b1s[64], b2s[32], w3s[32];
    __shared__ float b3s;
    __shared__ __align__(32) _Float16 h1s[WPB * 16 * 64];     // per-wave layer1 transpose scratch 16 KB
    __shared__ float facs[WPB * 16];

    const int tid = threadIdx.x;

    // Stage W1 (128x64) as B fragments: value = W1[K][n], K = 32c + 16*(j>>3) + 8*(lane>>4) + (j&7), n = 16nt + (lane&15)
    for (int idx = tid; idx < 4 * 4 * 32 * 16; idx += 256) {
        int j = idx & 15, lane = (idx >> 4) & 31, c = (idx >> 9) & 3, nt = idx >> 11;
        int K = 32 * c + 16 * (j >> 3) + 8 * (lane >> 4) + (j & 7);
        int n = 16 * nt + (lane & 15);
        w1B[idx] = (_Float16)W1[K * 64 + n];
    }
    // Stage W2 (64x32)
    for (int idx = tid; idx < 2 * 2 * 32 * 16; idx += 256) {
        int j = idx & 15, lane = (idx >> 4) & 31, c = (idx >> 9) & 1, nt = (idx >> 10) & 1;
        int K = 32 * c + 16 * (j >> 3) + 8 * (lane >> 4) + (j & 7);
        int n = 16 * nt + (lane & 15);
        w2B[idx] = (_Float16)W2[K * 32 + n];
    }
    if (tid < 64) b1s[tid] = b1[tid];
    if (tid < 32) { b2s[tid] = b2[tid]; w3s[tid] = W3[tid]; }
    if (tid == 0) b3s = b3[0];
    __syncthreads();

    const int wave = tid >> 5;
    const int lane = tid & 31;
    const int l15  = lane & 15;
    const int hgrp = lane >> 4;                 // 0 or 1 (which half of wave)
    const int wid  = blockIdx.x * WPB + wave;

    // Hoist loop-invariant operands
    v16h bw2[4];
#pragma unroll
    for (int nt = 0; nt < 2; ++nt)
#pragma unroll
        for (int c = 0; c < 2; ++c)
            bw2[nt * 2 + c] = *(const v16h*)&w2B[((nt * 2 + c) * 32 + lane) * 16];

    float bias1v[4];
#pragma unroll
    for (int nt = 0; nt < 4; ++nt) bias1v[nt] = b1s[16 * nt + l15];
    const float bias2v0 = b2s[l15], bias2v1 = b2s[16 + l15];
    const float w3a = w3s[l15], w3b = w3s[16 + l15];
    const float bias3 = b3s;

    _Float16* myh1 = &h1s[wave * 1024];
    float sumS = 0.0f, cntS = 0.0f;

    for (int tile = wid; tile < NTILES; tile += TOT_WAVES) {
        const int e = tile * 16 + l15;          // lanes L and L+16 own the same edge row
        const int srcN = (int)ei[e];
        const int tgtN = (int)ei[N_EDGES + e];
        const float* srow = feats + (long long)srcN * 64;
        const float* trow = feats + (long long)tgtN * 64;

        // Build A fragments for combined[16][128] (f16), 4 K-chunks of 32.
        v16h a[4];
#pragma unroll
        for (int c = 0; c < 4; ++c) {
            const float* base = (c < 2) ? srow : trow;
            const int kb0 = (c & 1) * 32 + 8 * hgrp;        // j=0..7  -> K = 32c + 8*hgrp + j
            v8h g0 = load8_cvt(base + kb0);                 // j=8..15 -> K = 32c + 16 + 8*hgrp + (j-8)
            v8h g1 = load8_cvt(base + kb0 + 16);
            a[c] = __builtin_shufflevector(g0, g1, 0,1,2,3,4,5,6,7,8,9,10,11,12,13,14,15);
        }

        // Layer 1: 16x128 @ 128x64 -> 16 WMMAs
        v8f acc[4] = {};
#pragma unroll
        for (int nt = 0; nt < 4; ++nt)
#pragma unroll
            for (int c = 0; c < 4; ++c) {
                v16h bf = *(const v16h*)&w1B[((nt * 4 + c) * 32 + lane) * 16];
                acc[nt] = __builtin_amdgcn_wmma_f32_16x16x32_f16(
                    false, a[c], false, bf, (short)0, acc[nt], false, false);
            }

        // Bias + ReLU; write h1 (f16) to per-wave LDS for D->A transpose
#pragma unroll
        for (int nt = 0; nt < 4; ++nt) {
            const int n = 16 * nt + l15;
            const float bb = bias1v[nt];
#pragma unroll
            for (int v = 0; v < 8; ++v) {
                float h = acc[nt][v] + bb;
                h = h > 0.0f ? h : 0.0f;
                myh1[(8 * hgrp + v) * 64 + n] = (_Float16)h;
            }
        }
        __builtin_amdgcn_wave_barrier();

        // Layer 2 A fragments from LDS
        v16h a2[2];
#pragma unroll
        for (int c = 0; c < 2; ++c) {
            const int kb0 = 32 * c + 8 * hgrp;
            v8h g0 = *(const v8h*)&myh1[l15 * 64 + kb0];
            v8h g1 = *(const v8h*)&myh1[l15 * 64 + kb0 + 16];
            a2[c] = __builtin_shufflevector(g0, g1, 0,1,2,3,4,5,6,7,8,9,10,11,12,13,14,15);
        }

        // Layer 2: 16x64 @ 64x32 -> 4 WMMAs
        v8f acc2[2] = {};
#pragma unroll
        for (int nt = 0; nt < 2; ++nt)
#pragma unroll
            for (int c = 0; c < 2; ++c)
                acc2[nt] = __builtin_amdgcn_wmma_f32_16x16x32_f16(
                    false, a2[c], false, bw2[nt * 2 + c], (short)0, acc2[nt], false, false);

        // Layer 3: bias+relu then dot with W3 (32x1); reduce over the 16 lanes of each half
        float part[8];
#pragma unroll
        for (int v = 0; v < 8; ++v) {
            float h0 = acc2[0][v] + bias2v0; h0 = h0 > 0.0f ? h0 : 0.0f;
            float h1 = acc2[1][v] + bias2v1; h1 = h1 > 0.0f ? h1 : 0.0f;
            part[v] = h0 * w3a + h1 * w3b;
        }
#pragma unroll
        for (int v = 0; v < 8; ++v) {
            part[v] += __shfl_xor(part[v], 1, 32);
            part[v] += __shfl_xor(part[v], 2, 32);
            part[v] += __shfl_xor(part[v], 4, 32);
            part[v] += __shfl_xor(part[v], 8, 32);
        }

        float sc[8], fac8[8];
#pragma unroll
        for (int v = 0; v < 8; ++v) {
            float s = 1.0f / (1.0f + expf(-(part[v] + bias3)));
            sc[v] = s;
            fac8[v] = (s < 0.7f) ? 0.05f * (1.0f - s) : 0.0f;
        }

        if (l15 == 0) {                         // lanes 0 (rows 0-7) and 16 (rows 8-15)
            float* sp = out + SCORE_OFF + tile * 16 + 8 * hgrp;
            v4f q0 = {sc[0], sc[1], sc[2], sc[3]};
            v4f q1 = {sc[4], sc[5], sc[6], sc[7]};
            *(v4f*)sp = q0;
            *((v4f*)sp + 1) = q1;
#pragma unroll
            for (int v = 0; v < 8; ++v) {
                sumS += sc[v];
                cntS += (sc[v] < 0.7f) ? 1.0f : 0.0f;
                facs[wave * 16 + 8 * hgrp + v] = fac8[v];
            }
        }
        __builtin_amdgcn_wave_barrier();
        const float fmine = facs[wave * 16 + l15];

        // Scatter 0.05*(1-s)*tanh(combined) back to src/tgt rows (L2-resident f32 atomics)
#pragma unroll
        for (int c = 0; c < 4; ++c) {
            float* obase = out + (long long)((c < 2) ? srcN : tgtN) * 64;
#pragma unroll
            for (int j = 0; j < 16; ++j) {
                const int K = 32 * c + ((j >> 3) ? 16 : 0) + 8 * hgrp + (j & 7);
                float t = tanhf((float)a[c][j]);
                atomicAdd(obase + (K & 63), fmine * t);
            }
        }
    }

    if (l15 == 0) {
        atomicAdd(&ws[0], sumS);
        atomicAdd(&ws[1], cntS);
    }
}

__global__ void vessel_final_kernel(const float* __restrict__ ws, float* __restrict__ out) {
    if (threadIdx.x == 0 && blockIdx.x == 0) {
        out[AVG_OFF]     = ws[0] / (float)N_EDGES;
        out[AVG_OFF + 1] = ws[1];
    }
}

extern "C" void kernel_launch(void* const* d_in, const int* in_sizes, int n_in,
                              void* d_out, int out_size, void* d_ws, size_t ws_size,
                              hipStream_t stream) {
    (void)in_sizes; (void)n_in; (void)out_size; (void)ws_size;
    const float*     feats = (const float*)d_in[0];
    const long long* ei    = (const long long*)d_in[1];   // int64 edge_index [2,E]
    const float*     W1    = (const float*)d_in[4];
    const float*     b1    = (const float*)d_in[5];
    const float*     W2    = (const float*)d_in[6];
    const float*     b2    = (const float*)d_in[7];
    const float*     W3    = (const float*)d_in[8];
    const float*     b3    = (const float*)d_in[9];
    float* out = (float*)d_out;
    float* ws  = (float*)d_ws;

    vessel_init_kernel<<<2048, 256, 0, stream>>>(feats, out, ws);
    vessel_edge_kernel<<<EBLOCKS, 256, 0, stream>>>(feats, ei, W1, b1, W2, b2, W3, b3, out, ws);
    vessel_final_kernel<<<1, 32, 0, stream>>>(ws, out);
}
